// ViTBlock_60224031424958
// MI455X (gfx1250) — matrix operand order repaired and checked by reading, not compiled, not run on hardware
//
#include <hip/hip_runtime.h>
#include <hip/hip_bf16.h>
#include <cstdint>

// ViT block for MI455X (gfx1250): WMMA f16 GEMMs + async global->LDS staging.
#define BATCH 16
#define CDIM  768
#define NTOK  1024
#define HEADS 12
#define HD    64
#define KSTEP 64
#define LSTR  72   // padded LDS row stride (halves) to avoid bank conflicts

typedef __attribute__((ext_vector_type(16))) _Float16 v16h;
typedef __attribute__((ext_vector_type(8)))  _Float16 v8h;
typedef __attribute__((ext_vector_type(8)))  float    v8f;

__device__ __forceinline__ v8f wmma_f16(v16h a, v16h b, v8f c) {
  // (neg_a, A, neg_b, B, c_mod, C, reuse_a, reuse_b)
  return __builtin_amdgcn_wmma_f32_16x16x32_f16(false, a, false, b, (short)0, c,
                                                false, false);
}

// Async copy 16 bytes global -> LDS (per lane), tracked by ASYNCcnt.
__device__ __forceinline__ void async_copy16(uint32_t lds_byte, const void* gaddr) {
  asm volatile("global_load_async_to_lds_b128 %0, %1, off"
               :: "v"(lds_byte), "v"(gaddr)
               : "memory");
}
__device__ __forceinline__ void wait_async0() {
  asm volatile("s_wait_asynccnt 0" ::: "memory");
}

// A-matrix 16x32 f16 fragment (ISA 7.12.2): lane<16 -> kb=k0, lane>=16 -> kb=k0+8;
// elements 0..7 = K kb..kb+7, elements 8..15 = K kb+16..kb+23.
__device__ __forceinline__ v16h frag_a(const _Float16* A, int lda, int row0,
                                       int k0, int lane) {
  int r  = row0 + (lane & 15);
  int kb = k0 + ((lane >> 4) << 3);
  const _Float16* p = A + (size_t)r * lda + kb;
  v8h lo = *(const v8h*)p;
  v8h hi = *(const v8h*)(p + 16);
  v16h o;
#pragma unroll
  for (int i = 0; i < 8; ++i) { o[i] = lo[i]; o[i + 8] = hi[i]; }
  return o;
}

// B-matrix 32x16 f16 fragment where B[k][n] = Wt[n][k] (Wt row-major [N,K]):
// lane holds column n = col0+(lane&15), kb = k0 + (lane>=16 ? 8 : 0); elements 0..7 = K kb..kb+7, 8..15 = K kb+16..kb+23.
__device__ __forceinline__ v16h frag_b(const _Float16* Wt, int ldw, int col0,
                                       int k0, int lane) {
  int c  = col0 + (lane & 15);
  int kb = k0 + ((lane >> 4) << 3);
  const _Float16* p = Wt + (size_t)c * ldw + kb;
  v8h lo = *(const v8h*)p;
  v8h hi = *(const v8h*)(p + 16);
  v16h o;
#pragma unroll
  for (int i = 0; i < 8; ++i) { o[i] = lo[i]; o[i + 8] = hi[i]; }
  return o;
}

__global__ void cvt_f32_to_f16_kernel(const float* __restrict__ s,
                                      _Float16* __restrict__ d, int n) {
  int i = blockIdx.x * blockDim.x + threadIdx.x;
  if (i < n) d[i] = (_Float16)s[i];
}

// LayerNorm over C=768. MODE 0: gather x[b,c,n] (strided), write xt fp32 + xn f16.
// MODE 1: read rows of fp32 [M,768], write xn f16 only.
template <int MODE>
__global__ __launch_bounds__(256) void ln_kernel(const float* __restrict__ src,
                                                 const float* __restrict__ gw,
                                                 const float* __restrict__ bw,
                                                 float* __restrict__ xt,
                                                 _Float16* __restrict__ xn) {
  const int row = blockIdx.x;              // b*1024 + n
  const int b = row >> 10, n = row & 1023;
  float val[3];
#pragma unroll
  for (int j = 0; j < 3; ++j) {
    int c = threadIdx.x + j * 256;
    val[j] = (MODE == 0) ? src[((size_t)b * CDIM + c) * NTOK + n]
                         : src[(size_t)row * CDIM + c];
  }
  float s1 = val[0] + val[1] + val[2];
  float s2 = val[0] * val[0] + val[1] * val[1] + val[2] * val[2];
#pragma unroll
  for (int off = 16; off > 0; off >>= 1) {
    s1 += __shfl_xor(s1, off, 32);
    s2 += __shfl_xor(s2, off, 32);
  }
  __shared__ float rs1[8], rs2[8];
  int wave = threadIdx.x >> 5;
  if ((threadIdx.x & 31) == 0) { rs1[wave] = s1; rs2[wave] = s2; }
  __syncthreads();
  float t1 = 0.f, t2 = 0.f;
#pragma unroll
  for (int w = 0; w < 8; ++w) { t1 += rs1[w]; t2 += rs2[w]; }
  float mean = t1 * (1.0f / CDIM);
  float var  = t2 * (1.0f / CDIM) - mean * mean;
  float rstd = rsqrtf(var + 1e-5f);
#pragma unroll
  for (int j = 0; j < 3; ++j) {
    int c = threadIdx.x + j * 256;
    float xv = val[j];
    if (MODE == 0) xt[(size_t)row * CDIM + c] = xv;
    xn[(size_t)row * CDIM + c] = (_Float16)((xv - mean) * rstd * gw[c] + bw[c]);
  }
}

// OUT[M, NOUT] = A[M, KD](f16) * W[NOUT, KD]^T(f16) + bias.
// Double-buffered async global->LDS tiles (A 64xKSTEP, B 128xKSTEP), WMMA from LDS.
// EPI 0: store f16. EPI 1: exact GELU, store f16.
// EPI 2: resid + x, store fp32. EPI 3: resid + x, store fp32 transposed [B,C,N].
// Block 256 thr = 8 waves (4x2). Tile 64x128; wave tile 16x64. Grid (M/64, NOUT/128).
template <int EPI, int KD, int NOUT>
__global__ __launch_bounds__(256) void gemm_kernel(
    const _Float16* __restrict__ A, const _Float16* __restrict__ W,
    const float* __restrict__ bias, _Float16* __restrict__ outh,
    float* __restrict__ outf, const float* __restrict__ resid) {
  const int lane = threadIdx.x & 31;
  const int wave = threadIdx.x >> 5;
  const int m_blk = blockIdx.x * 64;
  const int n_blk = blockIdx.y * 128;
  const int m0 = (wave & 3) * 16;   // within block tile
  const int n0 = (wave >> 2) * 64;

  __shared__ __align__(16) _Float16 ldsA[2][64 * LSTR];
  __shared__ __align__(16) _Float16 ldsB[2][128 * LSTR];

  // Cooperative async fill of one K-step tile: 512 + 1024 16B chunks, 256 threads.
  auto issue = [&](int buf, int k0) {
    const int t = threadIdx.x;
#pragma unroll
    for (int i = 0; i < 2; ++i) {               // A: 64 rows x 8 chunks
      int idx = t + i * 256;
      int r = idx >> 3, c = idx & 7;
      async_copy16((uint32_t)(size_t)&ldsA[buf][r * LSTR + c * 8],
                   A + (size_t)(m_blk + r) * KD + k0 + c * 8);
    }
#pragma unroll
    for (int i = 0; i < 4; ++i) {               // B: 128 rows x 8 chunks
      int idx = t + i * 256;
      int r = idx >> 3, c = idx & 7;
      async_copy16((uint32_t)(size_t)&ldsB[buf][r * LSTR + c * 8],
                   W + (size_t)(n_blk + r) * KD + k0 + c * 8);
    }
  };

  const v8f vz = {0, 0, 0, 0, 0, 0, 0, 0};
  v8f acc[4];
#pragma unroll
  for (int ct = 0; ct < 4; ++ct) acc[ct] = vz;

  issue(0, 0);
  wait_async0();
  __syncthreads();

  constexpr int STEPS = KD / KSTEP;
  for (int s = 0; s < STEPS; ++s) {
    if (s + 1 < STEPS) issue((s + 1) & 1, (s + 1) * KSTEP);
    const _Float16* tA = &ldsA[s & 1][0];
    const _Float16* tB = &ldsB[s & 1][0];
#pragma unroll
    for (int kk = 0; kk < KSTEP; kk += 32) {
      v16h a = frag_a(tA, LSTR, m0, kk, lane);
#pragma unroll
      for (int ct = 0; ct < 4; ++ct) {
        v16h b = frag_b(tB, LSTR, n0 + ct * 16, kk, lane);
        acc[ct] = wmma_f16(a, b, acc[ct]);
      }
    }
    if (s + 1 < STEPS) wait_async0();   // next buffer landed (per-wave)
    __syncthreads();                    // all waves done reading current buffer
  }

#pragma unroll
  for (int ct = 0; ct < 4; ++ct) {
    int col = n_blk + n0 + ct * 16 + (lane & 15);
    float bv = bias[col];
#pragma unroll
    for (int v = 0; v < 8; ++v) {
      int row = m_blk + m0 + v + ((lane >> 4) << 3);
      float x = acc[ct][v] + bv;
      if (EPI == 0) {
        outh[(size_t)row * NOUT + col] = (_Float16)x;
      } else if (EPI == 1) {
        float g = 0.5f * x * (1.0f + erff(x * 0.70710678118654752f));
        outh[(size_t)row * NOUT + col] = (_Float16)g;
      } else if (EPI == 2) {
        outf[(size_t)row * NOUT + col] = resid[(size_t)row * NOUT + col] + x;
      } else {
        int bb = row >> 10, nn = row & 1023;
        outf[((size_t)bb * CDIM + col) * NTOK + nn] =
            resid[(size_t)row * CDIM + col] + x;
      }
    }
  }
}

// Flash attention: 4 waves / block, block handles (b, head, 64 q-rows); wave = 16 rows.
__global__ __launch_bounds__(128) void attn_kernel(
    const _Float16* __restrict__ qbuf, const _Float16* __restrict__ kvbuf,
    const float* __restrict__ rpb, const int* __restrict__ rel,
    _Float16* __restrict__ obuf) {
  const int lane = threadIdx.x & 31;
  const int wave = threadIdx.x >> 5;
  const int qt = blockIdx.x & 15;
  const int h  = (blockIdx.x >> 4) % HEADS;
  const int b  = blockIdx.x / (16 * HEADS);
  const int qrow0 = qt * 64 + wave * 16;  // within N

  const _Float16* Q  = qbuf + (size_t)b * NTOK * CDIM + h * HD;          // lda 768
  const _Float16* Kp = kvbuf + (size_t)b * NTOK * (2 * CDIM) + h * HD;   // lda 1536
  const _Float16* Vp = Kp + CDIM;

  __shared__ __align__(16) _Float16 ldsP[4][16 * 64];  // per-wave P tile, A-layout
  __shared__ __align__(16) _Float16 ldsVt[64][72];     // V transposed [d][key], padded

  const v8f vz = {0, 0, 0, 0, 0, 0, 0, 0};
  v8f Oacc[4];
#pragma unroll
  for (int i = 0; i < 4; ++i) Oacc[i] = vz;
  float Mrow[8], Lrow[8];
#pragma unroll
  for (int v = 0; v < 8; ++v) { Mrow[v] = -1e30f; Lrow[v] = 0.0f; }

  v16h aq0 = frag_a(Q, CDIM, qrow0, 0, lane);
  v16h aq1 = frag_a(Q, CDIM, qrow0, 32, lane);

  for (int kt = 0; kt < 16; ++kt) {
    const int key0 = kt * 64;
    __syncthreads();  // protect ldsVt/ldsP reuse from previous iteration
    // ---- S = Q K^T (scaled) + bias ----
    v8f s[4];
#pragma unroll
    for (int ct = 0; ct < 4; ++ct) {
      v16h bk0 = frag_b(Kp, 2 * CDIM, key0 + ct * 16, 0, lane);
      v16h bk1 = frag_b(Kp, 2 * CDIM, key0 + ct * 16, 32, lane);
      v8f a = vz;
      a = wmma_f16(aq0, bk0, a);
      a = wmma_f16(aq1, bk1, a);
      s[ct] = a;
    }
#pragma unroll
    for (int ct = 0; ct < 4; ++ct) {
      int col = key0 + ct * 16 + (lane & 15);
#pragma unroll
      for (int v = 0; v < 8; ++v) {
        int row = qrow0 + v + ((lane >> 4) << 3);
        int ridx = rel[(size_t)row * NTOK + col];
        s[ct][v] = s[ct][v] * 0.125f + rpb[(size_t)ridx * HEADS + h];
      }
    }
    // ---- online softmax (row reductions within 16-lane halves) ----
#pragma unroll
    for (int v = 0; v < 8; ++v) {
      float mx = fmaxf(fmaxf(s[0][v], s[1][v]), fmaxf(s[2][v], s[3][v]));
#pragma unroll
      for (int off = 1; off < 16; off <<= 1) mx = fmaxf(mx, __shfl_xor(mx, off, 32));
      float mnew = fmaxf(Mrow[v], mx);
      float corr = __expf(Mrow[v] - mnew);
      Lrow[v] *= corr;
#pragma unroll
      for (int ct = 0; ct < 4; ++ct) Oacc[ct][v] *= corr;
      float rsum = 0.0f;
#pragma unroll
      for (int ct = 0; ct < 4; ++ct) {
        float p = __expf(s[ct][v] - mnew);
        s[ct][v] = p;
        rsum += p;
      }
#pragma unroll
      for (int off = 1; off < 16; off <<= 1) rsum += __shfl_xor(rsum, off, 32);
      Lrow[v] += rsum;
      Mrow[v] = mnew;
    }
    // ---- P -> LDS (C/D layout -> row-major [16 x 64]) ----
    _Float16* Pw = &ldsP[wave][0];
#pragma unroll
    for (int ct = 0; ct < 4; ++ct) {
      int c = ct * 16 + (lane & 15);
#pragma unroll
      for (int v = 0; v < 8; ++v) {
        int r = v + ((lane >> 4) << 3);
        Pw[r * 64 + c] = (_Float16)s[ct][v];
      }
    }
    // ---- V tile -> LDS transposed [d][key] (cooperative, vectorized global) ----
    for (int t = threadIdx.x; t < 512; t += 128) {
      int key = t >> 3, d0 = (t & 7) << 3;
      v8h vv = *(const v8h*)(Vp + (size_t)(key0 + key) * (2 * CDIM) + d0);
#pragma unroll
      for (int j = 0; j < 8; ++j) ldsVt[d0 + j][key] = vv[j];
    }
    __syncthreads();
    // ---- O += P * V ----
    v16h ap0 = frag_a(Pw, 64, 0, 0, lane);
    v16h ap1 = frag_a(Pw, 64, 0, 32, lane);
#pragma unroll
    for (int ctd = 0; ctd < 4; ++ctd) {
      v16h bv0 = frag_b(&ldsVt[0][0], 72, ctd * 16, 0, lane);
      v16h bv1 = frag_b(&ldsVt[0][0], 72, ctd * 16, 32, lane);
      Oacc[ctd] = wmma_f16(ap0, bv0, Oacc[ctd]);
      Oacc[ctd] = wmma_f16(ap1, bv1, Oacc[ctd]);
    }
  }
  // ---- finalize: O /= L, store f16 into [M, 768] at head columns ----
#pragma unroll
  for (int ct = 0; ct < 4; ++ct) {
    int col = h * HD + ct * 16 + (lane & 15);
#pragma unroll
    for (int v = 0; v < 8; ++v) {
      int row = b * NTOK + qrow0 + v + ((lane >> 4) << 3);
      obuf[(size_t)row * CDIM + col] = (_Float16)(Oacc[ct][v] / Lrow[v]);
    }
  }
}

extern "C" void kernel_launch(void* const* d_in, const int* in_sizes, int n_in,
                              void* d_out, int out_size, void* d_ws, size_t ws_size,
                              hipStream_t stream) {
  const float* x      = (const float*)d_in[0];
  const float* rpb    = (const float*)d_in[1];
  const float* g1     = (const float*)d_in[2];
  const float* b1     = (const float*)d_in[3];
  const float* q_w    = (const float*)d_in[4];
  const float* q_b    = (const float*)d_in[5];
  const float* kv_w   = (const float*)d_in[6];
  const float* kv_b   = (const float*)d_in[7];
  const float* proj_w = (const float*)d_in[8];
  const float* proj_b = (const float*)d_in[9];
  const float* g2     = (const float*)d_in[10];
  const float* b2     = (const float*)d_in[11];
  const float* fc1_w  = (const float*)d_in[12];
  const float* fc1_b  = (const float*)d_in[13];
  const float* fc2_w  = (const float*)d_in[14];
  const float* fc2_b  = (const float*)d_in[15];
  const int*   rel    = (const int*)d_in[16];
  float* out = (float*)d_out;

  char* ws = (char*)d_ws;
  size_t off = 0;
  auto take = [&](size_t bytes) -> void* {
    void* p = ws + off;
    off += (bytes + 255) & ~(size_t)255;
    return p;
  };
  const size_t M = (size_t)BATCH * NTOK;  // 16384 rows

  _Float16* wq    = (_Float16*)take((size_t)CDIM * CDIM * 2);
  _Float16* wkv   = (_Float16*)take((size_t)2 * CDIM * CDIM * 2);
  _Float16* wproj = (_Float16*)take((size_t)CDIM * CDIM * 2);
  _Float16* wfc1  = (_Float16*)take((size_t)4 * CDIM * CDIM * 2);
  _Float16* wfc2  = (_Float16*)take((size_t)CDIM * 4 * CDIM * 2);
  float*    xt    = (float*)take(M * CDIM * 4);
  _Float16* xnh   = (_Float16*)take(M * CDIM * 2);      // reused as o
  _Float16* qh    = (_Float16*)take(M * CDIM * 2);      // reused as xn2
  _Float16* kvh   = (_Float16*)take(M * 2 * CDIM * 2);  // reused as xt2 (fp32)
  _Float16* h1h   = (_Float16*)take(M * 4 * CDIM * 2);
  _Float16* oh    = xnh;
  _Float16* xn2h  = qh;
  float*    xt2   = (float*)kvh;

  auto cvt = [&](const float* s, _Float16* d, int n) {
    cvt_f32_to_f16_kernel<<<(n + 255) / 256, 256, 0, stream>>>(s, d, n);
  };
  cvt(q_w, wq, CDIM * CDIM);
  cvt(kv_w, wkv, 2 * CDIM * CDIM);
  cvt(proj_w, wproj, CDIM * CDIM);
  cvt(fc1_w, wfc1, 4 * CDIM * CDIM);
  cvt(fc2_w, wfc2, CDIM * 4 * CDIM);

  // LN1: x [B,C,H,W] -> xt fp32 [M,768] + xn f16
  ln_kernel<0><<<(unsigned)M, 256, 0, stream>>>(x, g1, b1, xt, xnh);
  // q = xn @ q_w^T + q_b      (f16 out [M,768])
  gemm_kernel<0, CDIM, CDIM><<<dim3(M / 64, CDIM / 128), 256, 0, stream>>>(
      xnh, wq, q_b, qh, nullptr, nullptr);
  // kv = xn @ kv_w^T + kv_b   (f16 out [M,1536]; k cols 0..767, v cols 768..1535)
  gemm_kernel<0, CDIM, 2 * CDIM><<<dim3(M / 64, (2 * CDIM) / 128), 256, 0, stream>>>(
      xnh, wkv, kv_b, kvh, nullptr, nullptr);
  // attention -> o (reuses xn buffer)
  attn_kernel<<<BATCH * HEADS * 16, 128, 0, stream>>>(qh, kvh, rpb, rel, oh);
  // xt2 = xt + o @ proj_w^T + proj_b  (fp32, reuses kv buffer)
  gemm_kernel<2, CDIM, CDIM><<<dim3(M / 64, CDIM / 128), 256, 0, stream>>>(
      oh, wproj, proj_b, nullptr, xt2, xt);
  // LN2 -> xn2 f16 (reuses q buffer)
  ln_kernel<1><<<(unsigned)M, 256, 0, stream>>>(xt2, g2, b2, nullptr, xn2h);
  // h1 = gelu(xn2 @ fc1_w^T + fc1_b)  (f16 [M,3072])
  gemm_kernel<1, CDIM, 4 * CDIM><<<dim3(M / 64, (4 * CDIM) / 128), 256, 0, stream>>>(
      xn2h, wfc1, fc1_b, h1h, nullptr, nullptr);
  // out = transpose(xt2 + h1 @ fc2_w^T + fc2_b) -> [B,C,H,W]
  gemm_kernel<3, 4 * CDIM, CDIM><<<dim3(M / 64, CDIM / 128), 256, 0, stream>>>(
      h1h, wfc2, fc2_b, nullptr, out, xt2);
}
